// GATSubNet_81269371175426
// MI455X (gfx1250) — matrix operand order repaired and checked by reading, not compiled, not run on hardware
//
#include <hip/hip_runtime.h>

typedef _Float16 half_t;
typedef __attribute__((ext_vector_type(16))) _Float16 v16h;
typedef __attribute__((ext_vector_type(8)))  float    v8f;

#define GAT_ALPHA 0.2f

__device__ __forceinline__ float lrelu(float v) { return v > 0.f ? v : GAT_ALPHA * v; }

constexpr int NN = 1024;   // nodes
constexpr int CC = 12;     // input channels
constexpr int HH = 8;      // heads (layer 1)
constexpr int DD = 64;     // per-head dim
constexpr int BATCH = 16;
constexpr int BH1 = BATCH * HH;   // 128

// ---------------------------------------------------------------------------
// K0: pack graph (int32 [N,N]) into bitmask [N][N/32] (wave32 ballot)
// ---------------------------------------------------------------------------
__global__ void pack_mask_kernel(const int* __restrict__ graph,
                                 unsigned* __restrict__ mbits) {
  int wid  = blockIdx.x * 8 + (threadIdx.x >> 5);   // wave id: N*32 total
  int lane = threadIdx.x & 31;
  int n = wid >> 5;
  int w = wid & 31;
  int m = w * 32 + lane;
  int pred = graph[(size_t)n * NN + m] > 0;
  unsigned long long bal = __ballot(pred);
  if (lane == 0) mbits[(size_t)n * 32 + w] = (unsigned)bal;
}

// ---------------------------------------------------------------------------
// K1: Wh1 = x @ W  (K=12, VALU), store Wh1^T as f16 [bh][D][N]; also s1,t1
// ---------------------------------------------------------------------------
__global__ void wh1_kernel(const float* __restrict__ x, const float* __restrict__ W,
                           const float* __restrict__ a_src, const float* __restrict__ a_dst,
                           half_t* __restrict__ whT1, float* __restrict__ s1,
                           float* __restrict__ t1) {
  int wid  = blockIdx.x * 8 + (threadIdx.x >> 5);
  int lane = threadIdx.x & 31;
  int bh = wid >> 10;
  int n  = wid & 1023;
  int b = bh >> 3, h = bh & 7;
  int d0 = lane, d1 = lane + 32;
  const float* xp = x + ((size_t)b * NN + n) * CC;
  const float* Wp = W + (size_t)h * CC * DD;
  float wh0 = 0.f, wh1 = 0.f;
#pragma unroll
  for (int c = 0; c < CC; ++c) {
    float xv = xp[c];
    wh0 += xv * Wp[c * DD + d0];
    wh1 += xv * Wp[c * DD + d1];
  }
  whT1[((size_t)bh * DD + d0) * NN + n] = (half_t)wh0;
  whT1[((size_t)bh * DD + d1) * NN + n] = (half_t)wh1;
  float sp = wh0 * a_src[h * DD + d0] + wh1 * a_src[h * DD + d1];
  float tp = wh0 * a_dst[h * DD + d0] + wh1 * a_dst[h * DD + d1];
  for (int off = 16; off; off >>= 1) {
    sp += __shfl_xor(sp, off);
    tp += __shfl_xor(tp, off);
  }
  if (lane == 0) { s1[(size_t)bh * NN + n] = sp; t1[(size_t)bh * NN + n] = tp; }
}

// ---------------------------------------------------------------------------
// K2: transpose+convert Wo [512,64] f32 -> WoT [64][512] f16
// ---------------------------------------------------------------------------
__global__ void wot_kernel(const float* __restrict__ Wo, half_t* __restrict__ woT) {
  int idx = blockIdx.x * blockDim.x + threadIdx.x;  // 64*512
  int o = idx & 63, k = idx >> 6;
  woT[(size_t)o * 512 + k] = (half_t)Wo[(size_t)k * 64 + o];
}

// ---------------------------------------------------------------------------
// K3: per-row online softmax stats (max, sum of exp) over masked lrelu(s+t)
// ---------------------------------------------------------------------------
__global__ void stats_kernel(const float* __restrict__ s, const float* __restrict__ t,
                             const unsigned* __restrict__ mbits,
                             float* __restrict__ rmax, float* __restrict__ rsum) {
  __shared__ float t_lds[NN];
  int row0 = blockIdx.x * 8;
  int bh = row0 >> 10;
  for (int i = threadIdx.x; i < NN; i += blockDim.x) t_lds[i] = t[(size_t)bh * NN + i];
  __syncthreads();
  int r    = row0 + (threadIdx.x >> 5);
  int lane = threadIdx.x & 31;
  int n = r & 1023;
  float sr = s[(size_t)bh * NN + n];
  const unsigned* mrow = mbits + (size_t)n * 32;
  float mx = -3.0e38f, sum = 0.f;
  for (int it = 0; it < 32; ++it) {
    unsigned w = mrow[it];
    float e = lrelu(sr + t_lds[it * 32 + lane]);
    if ((w >> lane) & 1u) {
      if (e > mx) { sum = sum * __expf(mx - e) + 1.f; mx = e; }
      else        { sum += __expf(e - mx); }
    }
  }
  for (int off = 16; off; off >>= 1) {
    float om = __shfl_xor(mx, off);
    float osum = __shfl_xor(sum, off);
    float M = fmaxf(mx, om);
    sum = sum * __expf(mx - M) + osum * __expf(om - M);
    mx = M;
  }
  if (lane == 0) { rmax[r] = mx; rsum[r] = sum; }
}

// ---------------------------------------------------------------------------
// A-fragment builder: P = exp(lrelu(s_n + t_m) - rowmax), masked, f16.
// f16 16x32 A layout: lane row = lane&15; lanes<16 K in {0..7,16..23},
// lanes>=16 K in {8..15,24..31}; element pair 2v,2v+1 per VGPR.
// ---------------------------------------------------------------------------
__device__ __forceinline__ v16h build_A(float sr, float rm, unsigned mw,
                                        const float* t_lds, int m0, int kb) {
  union { v16h v; half_t e[16]; } A;
#pragma unroll
  for (int v = 0; v < 8; ++v) {
    int k = (v < 4) ? (kb + 2 * v) : (kb + 16 + 2 * (v - 4));
    float e0 = lrelu(sr + t_lds[m0 + k]);
    float e1 = lrelu(sr + t_lds[m0 + k + 1]);
    float p0 = ((mw >> k) & 1u)       ? __expf(e0 - rm) : 0.f;
    float p1 = ((mw >> (k + 1)) & 1u) ? __expf(e1 - rm) : 0.f;
    A.e[2 * v]     = (half_t)p0;
    A.e[2 * v + 1] = (half_t)p1;
  }
  return A.v;
}

// ---------------------------------------------------------------------------
// K4: fused masked-softmax attention * V via WMMA f16.
//     Each wave owns TWO 16-row n-tiles (32 rows) sharing the B fragments;
//     loops m in 32-wide K-tiles. B loads issued first so the ~50 VALU ops
//     of the A build hide their latency.
// ---------------------------------------------------------------------------
template <int HEADS, int OUTSTRIDE, int TILES_PER_BH, bool OUT_HALF>
__global__ void attn_kernel(const float* __restrict__ s, const float* __restrict__ t,
                            const float* __restrict__ rmax, const float* __restrict__ rsum,
                            const half_t* __restrict__ whT, const unsigned* __restrict__ mbits,
                            half_t* __restrict__ outH, float* __restrict__ outF) {
  __shared__ float t_lds[NN];
  int bh   = blockIdx.x / TILES_PER_BH;
  int tile = blockIdx.x % TILES_PER_BH;
  int wave = threadIdx.x >> 5, lane = threadIdx.x & 31;
  for (int i = threadIdx.x; i < NN; i += blockDim.x) t_lds[i] = t[(size_t)bh * NN + i];
  __syncthreads();

  int base  = tile * 256 + wave * 32;          // 32 rows per wave (2 A tiles)
  int col   = lane & 15;
  int half8 = (lane < 16) ? 0 : 8;
  int kb    = (lane < 16) ? 0 : 8;             // A-fragment K base
  int r0 = base + col;                         // tile-0 A row
  int r1 = base + 16 + col;                    // tile-1 A row
  float sr0 = s[(size_t)bh * NN + r0];
  float rm0 = rmax[(size_t)bh * NN + r0];
  float sr1 = s[(size_t)bh * NN + r1];
  float rm1 = rmax[(size_t)bh * NN + r1];
  const half_t* whbh = whT + (size_t)bh * DD * NN;

  v8f acc[2][4] = {};

  for (int m0 = 0; m0 < NN; m0 += 32) {
    // ---- B fragments first (8x b128 in flight while A is computed) ----
    int kr = m0 + ((lane < 16) ? 0 : 8);
    union { v16h v; uint4 u[2]; } Bf[4];
#pragma unroll
    for (int j = 0; j < 4; ++j) {
      const half_t* bp = whbh + (size_t)(j * 16 + col) * NN + kr;
      Bf[j].u[0] = *(const uint4*)bp;
      Bf[j].u[1] = *(const uint4*)(bp + 16);
    }
    // ---- A fragments (exp/cvt VALU overlaps the loads) ----
    unsigned mw0 = mbits[(size_t)r0 * 32 + (m0 >> 5)];
    unsigned mw1 = mbits[(size_t)r1 * 32 + (m0 >> 5)];
    v16h A0 = build_A(sr0, rm0, mw0, t_lds, m0, kb);
    v16h A1 = build_A(sr1, rm1, mw1, t_lds, m0, kb);
    // ---- 8 WMMAs ----
#pragma unroll
    for (int j = 0; j < 4; ++j) {
      acc[0][j] = __builtin_amdgcn_wmma_f32_16x16x32_f16(
          false, A0, false, Bf[j].v, (short)0, acc[0][j], false, false);
      acc[1][j] = __builtin_amdgcn_wmma_f32_16x16x32_f16(
          false, A1, false, Bf[j].v, (short)0, acc[1][j], false, false);
    }
  }

  // ---- normalize by rowsum, lrelu, store ----
  int b = bh / HEADS;
  int colOff = (bh % HEADS) * 64;
#pragma unroll
  for (int tl = 0; tl < 2; ++tl) {
    float rinv[8];
#pragma unroll
    for (int v = 0; v < 8; ++v) {
      float z = rsum[(size_t)bh * NN + base + tl * 16 + half8 + v];
      rinv[v] = z > 0.f ? 1.f / z : 0.f;
    }
#pragma unroll
    for (int j = 0; j < 4; ++j) {
      union { v8f v; float f[8]; } Ac; Ac.v = acc[tl][j];
#pragma unroll
      for (int v = 0; v < 8; ++v) {
        int row = base + tl * 16 + half8 + v;
        float val = lrelu(Ac.f[v] * rinv[v]);
        size_t idx = ((size_t)(b * NN + row)) * OUTSTRIDE + colOff + j * 16 + col;
        if constexpr (OUT_HALF) outH[idx] = (half_t)val;
        else                    outF[idx] = val;
      }
    }
  }
}

// ---------------------------------------------------------------------------
// K5: Wh2 = h1 [B*N,512] @ WoT^T -> f32 [B*N,64] + f16 Wh2^T [b][64][N], WMMA
// ---------------------------------------------------------------------------
__global__ void wh2_kernel(const half_t* __restrict__ h1, const half_t* __restrict__ woT,
                           float* __restrict__ wh2, half_t* __restrict__ whT2) {
  int wid  = blockIdx.x * 8 + (threadIdx.x >> 5);  // 0..(B*N/16 - 1)
  int lane = threadIdx.x & 31;
  int rowTile = wid * 16;
  int col   = lane & 15;
  int kb    = (lane < 16) ? 0 : 8;
  int half8 = (lane < 16) ? 0 : 8;
  const half_t* arow = h1 + (size_t)(rowTile + col) * 512;
  v8f acc[4] = {};
  for (int k0 = 0; k0 < 512; k0 += 32) {
    int kr = k0 + ((lane < 16) ? 0 : 8);
    // all loads first (A + 4 B fragments), then the WMMAs
    union { v16h v; uint4 u[2]; } A;
    A.u[0] = *(const uint4*)(arow + k0 + kb);
    A.u[1] = *(const uint4*)(arow + k0 + kb + 16);
    union { v16h v; uint4 u[2]; } Bf[4];
#pragma unroll
    for (int j = 0; j < 4; ++j) {
      const half_t* bp = woT + (size_t)(j * 16 + col) * 512 + kr;
      Bf[j].u[0] = *(const uint4*)bp;
      Bf[j].u[1] = *(const uint4*)(bp + 16);
    }
#pragma unroll
    for (int j = 0; j < 4; ++j) {
      acc[j] = __builtin_amdgcn_wmma_f32_16x16x32_f16(
          false, A.v, false, Bf[j].v, (short)0, acc[j], false, false);
    }
  }
  int b = rowTile >> 10;
#pragma unroll
  for (int j = 0; j < 4; ++j) {
    union { v8f v; float f[8]; } Ac; Ac.v = acc[j];
#pragma unroll
    for (int v = 0; v < 8; ++v) {
      int row = rowTile + half8 + v;      // global row in [0, B*N)
      int o = j * 16 + col;
      float val = Ac.f[v];
      wh2[(size_t)row * 64 + o] = val;
      whT2[((size_t)b * 64 + o) * NN + (row & 1023)] = (half_t)val;
    }
  }
}

// ---------------------------------------------------------------------------
// K6: s2/t2 = Wh2 . ao_src/ao_dst ; one wave per row
// ---------------------------------------------------------------------------
__global__ void st2_kernel(const float* __restrict__ wh2, const float* __restrict__ ao_src,
                           const float* __restrict__ ao_dst,
                           float* __restrict__ s2, float* __restrict__ t2) {
  int wid  = blockIdx.x * 8 + (threadIdx.x >> 5);
  int lane = threadIdx.x & 31;
  float w0 = wh2[(size_t)wid * 64 + lane];
  float w1 = wh2[(size_t)wid * 64 + lane + 32];
  float sp = w0 * ao_src[lane] + w1 * ao_src[lane + 32];
  float tp = w0 * ao_dst[lane] + w1 * ao_dst[lane + 32];
  for (int off = 16; off; off >>= 1) {
    sp += __shfl_xor(sp, off);
    tp += __shfl_xor(tp, off);
  }
  if (lane == 0) { s2[wid] = sp; t2[wid] = tp; }
}

// ---------------------------------------------------------------------------
extern "C" void kernel_launch(void* const* d_in, const int* in_sizes, int n_in,
                              void* d_out, int out_size, void* d_ws, size_t ws_size,
                              hipStream_t stream) {
  (void)in_sizes; (void)n_in; (void)out_size; (void)ws_size;
  const float* x      = (const float*)d_in[0];
  const int*   graph  = (const int*)d_in[1];
  const float* W      = (const float*)d_in[2];
  const float* a_src  = (const float*)d_in[3];
  const float* a_dst  = (const float*)d_in[4];
  const float* Wo     = (const float*)d_in[5];
  const float* ao_src = (const float*)d_in[6];
  const float* ao_dst = (const float*)d_in[7];
  float* out = (float*)d_out;

  char* ws = (char*)d_ws;
  size_t off = 0;
  auto alloc = [&](size_t bytes) -> char* {
    char* p = ws + off;
    off += (bytes + 255) & ~(size_t)255;
    return p;
  };
  unsigned* mbits = (unsigned*)alloc((size_t)NN * 32 * 4);          // 128 KB
  float* s1    = (float*)alloc((size_t)BH1 * NN * 4);               // 512 KB
  float* t1    = (float*)alloc((size_t)BH1 * NN * 4);
  float* rmax1 = (float*)alloc((size_t)BH1 * NN * 4);
  float* rsum1 = (float*)alloc((size_t)BH1 * NN * 4);
  half_t* whT1 = (half_t*)alloc((size_t)BH1 * DD * NN * 2);         // 16.8 MB
  half_t* h1   = (half_t*)alloc((size_t)BATCH * NN * 512 * 2);      // 16.8 MB
  half_t* woT  = (half_t*)alloc((size_t)64 * 512 * 2);              // 64 KB
  float* wh2p  = (float*)alloc((size_t)BATCH * NN * 64 * 4);        // 4 MB
  half_t* whT2 = (half_t*)alloc((size_t)BATCH * 64 * NN * 2);       // 2 MB
  float* s2    = (float*)alloc((size_t)BATCH * NN * 4);
  float* t2    = (float*)alloc((size_t)BATCH * NN * 4);
  float* rmax2 = (float*)alloc((size_t)BATCH * NN * 4);
  float* rsum2 = (float*)alloc((size_t)BATCH * NN * 4);

  // mask bitmask
  pack_mask_kernel<<<NN * 32 / 8, 256, 0, stream>>>(graph, mbits);
  // layer-1 feature transform + scores
  wh1_kernel<<<BH1 * NN / 8, 256, 0, stream>>>(x, W, a_src, a_dst, whT1, s1, t1);
  // Wo transpose/convert
  wot_kernel<<<(64 * 512) / 256, 256, 0, stream>>>(Wo, woT);
  // layer-1 softmax stats
  stats_kernel<<<BH1 * NN / 8, 256, 0, stream>>>(s1, t1, mbits, rmax1, rsum1);
  // layer-1 attention (WMMA) -> h1 (f16, lrelu applied)
  attn_kernel<HH, 512, NN / 256, true>
      <<<BH1 * (NN / 256), 256, 0, stream>>>(s1, t1, rmax1, rsum1, whT1, mbits, h1, nullptr);
  // layer-2 feature transform (WMMA, K=512)
  wh2_kernel<<<(BATCH * NN / 16) / 8, 256, 0, stream>>>(h1, woT, wh2p, whT2);
  // layer-2 scores
  st2_kernel<<<BATCH * NN / 8, 256, 0, stream>>>(wh2p, ao_src, ao_dst, s2, t2);
  // layer-2 softmax stats
  stats_kernel<<<BATCH * NN / 8, 256, 0, stream>>>(s2, t2, mbits, rmax2, rsum2);
  // layer-2 attention (WMMA) -> out (f32, lrelu applied)
  attn_kernel<1, 64, NN / 256, false>
      <<<BATCH * (NN / 256), 256, 0, stream>>>(s2, t2, rmax2, rsum2, whT2, mbits, nullptr, out);
}
